// JT_Channel_76819785056895
// MI455X (gfx1250) — compile-verified, matrix operand order checked by reading, not run on hardware
//
#include <hip/hip_runtime.h>
#include <hip/hip_bf16.h>
#include <cstdint>

#define N_NODES 100000
#define N_EDGES 200000
#define N_GRAPH 2000
#define CEILDIV(a,b) (((a)+(b)-1)/(b))

typedef __attribute__((ext_vector_type(16))) _Float16 v16h;
typedef __attribute__((ext_vector_type(8)))  float    v8f;
typedef _Float16 h16;

__device__ __forceinline__ float leakyf(float x){ return x > 0.f ? x : 0.01f * x; }
__device__ __forceinline__ float eluf(float x){ return x > 0.f ? x : (expf(x) - 1.f); }
__device__ __forceinline__ float sigm(float x){ return 1.f / (1.f + expf(-x)); }

__device__ __forceinline__ unsigned fkey(float f){
  int i = __float_as_int(f);
  return (i >= 0) ? ((unsigned)i ^ 0x80000000u) : ~(unsigned)i;
}
__device__ __forceinline__ float kfloat(unsigned k){
  int i = (k & 0x80000000u) ? (int)(k ^ 0x80000000u) : (int)(~k);
  return __int_as_float(i);
}

// CDNA5 async copy: 16B global -> LDS, tracked by ASYNCcnt (ISA 15.18.3 op 98)
__device__ __forceinline__ void async_copy_b128(const h16* lds_dst, const h16* gsrc){
  unsigned lds_off = (unsigned)(uintptr_t)lds_dst;   // low 32 bits of flat LDS addr = LDS offset
  asm volatile("global_load_async_to_lds_b128 %0, %1, off"
               :: "v"(lds_off), "v"(gsrc) : "memory");
}

// ---------------------------------------------------------------------------
// WMMA GEMM (fp16 operands, fp32 accum):
//   C16[M,N] = act(A16[M,K] @ Bt16[N,K]^T + bias)
// Preconditions: K % 32 == 0, N % 64 == 0, lda/ldb % 8 == 0 (16B rows).
// Tile 128x64 per 256-thr block (8 wave32); K step 32.
// Double-buffered async-to-LDS staging: copy tile k+1 while WMMA-ing tile k.
// ---------------------------------------------------------------------------
#define BM 128
#define BN 64
#define BK 32
#define LDST 40   // LDS row stride in halves (80B: keeps 16B chunk alignment)

__device__ __forceinline__ void stage_tile(const h16* __restrict__ A,
                                           const h16* __restrict__ Bt,
                                           h16* as, h16* bs,
                                           int tid, int rowBase, int colBase,
                                           int M, int lda, int ldb, int k0)
{
  // A tile 128x32h = 512 x 16B chunks, 2 per thread
  #pragma unroll
  for (int i = 0; i < 2; ++i) {
    int q  = tid + i * 256;
    int r  = q >> 2;
    int c8 = (q & 3) * 8;
    int gr = rowBase + r; gr = gr < M ? gr : M - 1;   // clamp: rows >= M never stored
    async_copy_b128(as + r * LDST + c8, A + (size_t)gr * lda + k0 + c8);
  }
  // B tile 64x32h = 256 x 16B chunks, 1 per thread
  {
    int r  = tid >> 2;
    int c8 = (tid & 3) * 8;
    int gn = colBase + r;                              // N % 64 == 0 -> in range
    async_copy_b128(bs + r * LDST + c8, Bt + (size_t)gn * ldb + k0 + c8);
  }
}

template<int ACT, bool HAS_BIAS>
__global__ __launch_bounds__(256)
void gemm_wmma_h(const h16* __restrict__ A, const h16* __restrict__ Bt,
                 const float* __restrict__ bias, h16* __restrict__ C,
                 int M, int N, int K, int lda, int ldb, int ldc)
{
  __shared__ h16 As[2][BM * LDST];
  __shared__ h16 Bs[2][BN * LDST];
  const int tid  = threadIdx.x;
  const int lane = tid & 31;
  const int wave = tid >> 5;
  const int rowBase = blockIdx.y * BM;
  const int colBase = blockIdx.x * BN;

  v8f acc0 = {0,0,0,0,0,0,0,0};
  v8f acc1 = acc0, acc2 = acc0, acc3 = acc0;

  const int khalf = lane >> 4;
  const int mlane = lane & 15;

  // prologue: stage first tile (3 async ops per wave)
  stage_tile(A, Bt, As[0], Bs[0], tid, rowBase, colBase, M, lda, ldb, 0);

  int buf = 0;
  for (int k0 = 0; k0 < K; k0 += BK) {
    const bool more = (k0 + BK) < K;
    if (more)
      stage_tile(A, Bt, As[buf ^ 1], Bs[buf ^ 1], tid, rowBase, colBase, M, lda, ldb, k0 + BK);

    // async ops complete in order: <=3 outstanding means the older tile landed
    if (more) asm volatile("s_wait_asynccnt 0x3" ::: "memory");
    else      asm volatile("s_wait_asynccnt 0x0" ::: "memory");
    __syncthreads();

    const h16* as = As[buf];
    const h16* bs = Bs[buf];
    union Frag { unsigned u[8]; v16h v; };
    Frag af, bf0, bf1, bf2, bf3;
    const int mrow = wave * 16 + mlane;
    #pragma unroll
    for (int v = 0; v < 8; ++v) {
      int kk = ((v >> 2) << 4) + (khalf << 3) + ((v & 3) << 1);
      af.u[v]  = *(const unsigned*)(as + mrow * LDST + kk);
      bf0.u[v] = *(const unsigned*)(bs + ( 0 + mlane) * LDST + kk);
      bf1.u[v] = *(const unsigned*)(bs + (16 + mlane) * LDST + kk);
      bf2.u[v] = *(const unsigned*)(bs + (32 + mlane) * LDST + kk);
      bf3.u[v] = *(const unsigned*)(bs + (48 + mlane) * LDST + kk);
    }
    acc0 = __builtin_amdgcn_wmma_f32_16x16x32_f16(false, af.v, false, bf0.v, (short)0, acc0, false, false);
    acc1 = __builtin_amdgcn_wmma_f32_16x16x32_f16(false, af.v, false, bf1.v, (short)0, acc1, false, false);
    acc2 = __builtin_amdgcn_wmma_f32_16x16x32_f16(false, af.v, false, bf2.v, (short)0, acc2, false, false);
    acc3 = __builtin_amdgcn_wmma_f32_16x16x32_f16(false, af.v, false, bf3.v, (short)0, acc3, false, false);
    __syncthreads();   // reads done (waves waited dscnt before WMMA) -> buf^1 reusable
    buf ^= 1;
  }

  v8f accs[4] = {acc0, acc1, acc2, acc3};
  #pragma unroll
  for (int nt = 0; nt < 4; ++nt) {
    const int gc = colBase + nt * 16 + mlane;
    float bv = 0.f;
    if (HAS_BIAS) bv = bias[gc];
    #pragma unroll
    for (int v = 0; v < 8; ++v) {
      int gr = rowBase + wave * 16 + (khalf << 3) + v;
      if (gr < M) {
        float val = accs[nt][v] + bv;
        if (ACT == 1)      val = leakyf(val);
        else if (ACT == 2) val = fmaxf(val, 0.f);
        C[(size_t)gr * ldc + gc] = (h16)val;
      }
    }
  }
}

// ---------------------------------------------------------------------------
// Weight prep: fp32 -> fp16, optional transpose, K padded to Kp with zeros
// ---------------------------------------------------------------------------
__global__ void conv_transpose_h(const float* __restrict__ in, h16* __restrict__ out,
                                 int K, int N, int Kp){
  int i = blockIdx.x * blockDim.x + threadIdx.x;
  if (i >= N * Kp) return;
  int n = i / Kp, k = i - n * Kp;
  out[i] = (k < K) ? (h16)in[(size_t)k * N + n] : (h16)0.f;
}
__global__ void conv_pad_h(const float* __restrict__ in, h16* __restrict__ out,
                           int K, long N, int Kp){
  long i = (long)blockIdx.x * blockDim.x + threadIdx.x;
  if (i >= N * Kp) return;
  long n = i / Kp; int k = (int)(i - n * Kp);
  out[i] = (k < K) ? (h16)in[n * K + k] : (h16)0.f;
}

// ---------------------------------------------------------------------------
// Elementwise / scatter kernels
// ---------------------------------------------------------------------------
__global__ void fill_f32(float* p, float v, int n){
  int i = blockIdx.x * blockDim.x + threadIdx.x; if (i < n) p[i] = v;
}
__global__ void fill_u32(unsigned* p, unsigned v, int n){
  int i = blockIdx.x * blockDim.x + threadIdx.x; if (i < n) p[i] = v;
}

__global__ void bn_leaky_concat(const h16* __restrict__ xe, const float* __restrict__ xm,
                                const float* g, const float* b, const float* mean,
                                const float* var, h16* __restrict__ xcat, int n){
  int i = blockIdx.x * blockDim.x + threadIdx.x;
  if (i >= n * 128) return;
  int row = i >> 7, c = i & 127;
  float v = (float)xe[i];
  v = (v - mean[c]) * rsqrtf(var[c] + 1e-5f) * g[c] + b[c];
  xcat[(size_t)row * 256 + 128 + c] = (h16)leakyf(v);
  xcat[(size_t)row * 256 + c] = (h16)xm[i];
}

__global__ void edge_concat(const h16* __restrict__ x, const float* __restrict__ ea,
                            const int* __restrict__ src, h16* __restrict__ out, int E){
  int i = blockIdx.x * blockDim.x + threadIdx.x;
  if (i >= E * 160) return;
  int e = i / 160, c = i - e * 160;
  h16 v = (h16)0.f;
  if (c < 128)      v = x[(size_t)src[e] * 128 + c];
  else if (c < 144) v = (h16)ea[(size_t)e * 16 + (c - 128)];
  out[i] = v;
}

__global__ void rowdot(const h16* __restrict__ X, const float* __restrict__ w,
                       float* __restrict__ out, int rows){
  int r = blockIdx.x * blockDim.x + threadIdx.x;
  if (r >= rows) return;
  const h16* p = X + (size_t)r * 128;
  float s = 0.f;
  #pragma unroll
  for (int c = 0; c < 128; ++c) s += (float)p[c] * w[c];
  out[r] = s;
}

__global__ void combine_gate(float* __restrict__ logit, const float* __restrict__ nd,
                             const int* __restrict__ dst, int E){
  int e = blockIdx.x * blockDim.x + threadIdx.x;
  if (e < E) logit[e] = leakyf(logit[e] + nd[dst[e]]);
}
__global__ void combine_gat(const float* __restrict__ ns, const float* __restrict__ nd,
                            const int* __restrict__ src, const int* __restrict__ dst,
                            float* __restrict__ logit, int E){
  int e = blockIdx.x * blockDim.x + threadIdx.x;
  if (e < E) logit[e] = leakyf(ns[src[e]] + nd[dst[e]]);
}

__global__ void seg_max(const float* __restrict__ logit, const int* __restrict__ dst,
                        unsigned* __restrict__ keys, int E){
  int e = blockIdx.x * blockDim.x + threadIdx.x;
  if (e < E) atomicMax(&keys[dst[e]], fkey(logit[e]));
}
__global__ void seg_expsum(const float* __restrict__ logit, const int* __restrict__ dst,
                           const unsigned* __restrict__ keys, float* __restrict__ ex,
                           float* __restrict__ sum, int E){
  int e = blockIdx.x * blockDim.x + threadIdx.x;
  if (e >= E) return;
  float m = kfloat(keys[dst[e]]);
  float v = expf(logit[e] - m);
  ex[e] = v;
  atomicAdd(&sum[dst[e]], v);
}
__global__ void seg_norm(float* __restrict__ ex, const float* __restrict__ sum,
                         const int* __restrict__ dst, int E){
  int e = blockIdx.x * blockDim.x + threadIdx.x;
  if (e < E) ex[e] = ex[e] / (sum[dst[e]] + 1e-16f);
}

__global__ void scatter_wsum(const h16* __restrict__ Y, const float* __restrict__ a,
                             const int* __restrict__ dst, float* __restrict__ H, int E){
  int i = blockIdx.x * blockDim.x + threadIdx.x;
  if (i >= E * 128) return;
  int e = i >> 7, c = i & 127;
  atomicAdd(&H[(size_t)dst[e] * 128 + c], (float)Y[i] * a[e]);
}
__global__ void scatter_gat(const h16* __restrict__ XT, const float* __restrict__ a,
                            const int* __restrict__ src, const int* __restrict__ dst,
                            float* __restrict__ H, int E){
  int i = blockIdx.x * blockDim.x + threadIdx.x;
  if (i >= E * 128) return;
  int e = i >> 7, c = i & 127;
  atomicAdd(&H[(size_t)dst[e] * 128 + c], (float)XT[(size_t)src[e] * 128 + c] * a[e]);
}

__global__ void bias_elu_h(const float* __restrict__ H, const float* __restrict__ b,
                           h16* __restrict__ out, int n){
  int i = blockIdx.x * blockDim.x + threadIdx.x;
  if (i < n) { int c = i & 127; out[i] = (h16)eluf(H[i] + b[c]); }
}

__global__ void gru_relu(const h16* __restrict__ gi, const h16* __restrict__ gh,
                         const h16* __restrict__ hprev, h16* __restrict__ out, int n){
  int i = blockIdx.x * blockDim.x + threadIdx.x;
  if (i >= n) return;
  int row = i >> 7, c = i & 127;
  size_t b = (size_t)row * 384;
  float r = sigm((float)gi[b + c]       + (float)gh[b + c]);
  float z = sigm((float)gi[b + 128 + c] + (float)gh[b + 128 + c]);
  float nn = tanhf((float)gi[b + 256 + c] + r * (float)gh[b + 256 + c]);
  float h = (float)hprev[i];
  out[i] = (h16)fmaxf((1.f - z) * nn + z * h, 0.f);
}

__global__ void pool_add(const h16* __restrict__ X, const int* __restrict__ batch,
                         float* __restrict__ pool, int n){
  int i = blockIdx.x * blockDim.x + threadIdx.x;
  if (i >= n) return;
  int row = i >> 7, c = i & 127;
  atomicAdd(&pool[(size_t)batch[row] * 128 + c], (float)X[i]);
}

__global__ void finalize_mean_relu(const float* __restrict__ pool, float* __restrict__ out, int n){
  int i = blockIdx.x * blockDim.x + threadIdx.x;
  if (i < n) out[i] = fmaxf(pool[i] * 0.25f, 0.f);
}

// ---------------------------------------------------------------------------
// Host orchestration
// ---------------------------------------------------------------------------
static inline void gemm(const h16* A, const h16* Bt, const float* bias, h16* C,
                        int M, int N, int K, int lda, int ldb, int ldc,
                        int act, hipStream_t s){
  dim3 grid(CEILDIV(N, BN), CEILDIV(M, BM));
  if (bias) {
    if (act == 0)      gemm_wmma_h<0,true ><<<grid,256,0,s>>>(A,Bt,bias,C,M,N,K,lda,ldb,ldc);
    else if (act == 1) gemm_wmma_h<1,true ><<<grid,256,0,s>>>(A,Bt,bias,C,M,N,K,lda,ldb,ldc);
    else               gemm_wmma_h<2,true ><<<grid,256,0,s>>>(A,Bt,bias,C,M,N,K,lda,ldb,ldc);
  } else {
    if (act == 0)      gemm_wmma_h<0,false><<<grid,256,0,s>>>(A,Bt,bias,C,M,N,K,lda,ldb,ldc);
    else if (act == 1) gemm_wmma_h<1,false><<<grid,256,0,s>>>(A,Bt,bias,C,M,N,K,lda,ldb,ldc);
    else               gemm_wmma_h<2,false><<<grid,256,0,s>>>(A,Bt,bias,C,M,N,K,lda,ldb,ldc);
  }
}

extern "C" void kernel_launch(void* const* d_in, const int* in_sizes, int n_in,
                              void* d_out, int out_size, void* d_ws, size_t ws_size,
                              hipStream_t stream) {
  const int N = N_NODES, E = N_EDGES, G = N_GRAPH;

  const float* x         = (const float*)d_in[0];
  const float* x_motif   = (const float*)d_in[1];
  const float* edge_attr = (const float*)d_in[2];
  const float* emb_W     = (const float*)d_in[3];
  const float* emb_b     = (const float*)d_in[4];
  const float* bn_gamma  = (const float*)d_in[5];
  const float* bn_beta   = (const float*)d_in[6];
  const float* bn_mean   = (const float*)d_in[7];
  const float* bn_var    = (const float*)d_in[8];
  const float* proj_W    = (const float*)d_in[9];
  const float* proj_b    = (const float*)d_in[10];
  const float* lin1_W    = (const float*)d_in[11];
  const float* lin1_b    = (const float*)d_in[12];
  const float* gateL1    = (const float*)d_in[13];
  const float* gateL2    = (const float*)d_in[14];
  const float* gate_al   = (const float*)d_in[15];
  const float* gate_ar   = (const float*)d_in[16];
  const float* gate_bias = (const float*)d_in[17];
  const float* g0_Wih    = (const float*)d_in[18];
  const float* g0_Whh    = (const float*)d_in[19];
  const float* g0_bih    = (const float*)d_in[20];
  const float* g0_bhh    = (const float*)d_in[21];
  const float* gat_W     = (const float*)d_in[22];
  const float* gat_as    = (const float*)d_in[23];
  const float* gat_ad    = (const float*)d_in[24];
  const float* gat_bias  = (const float*)d_in[25];
  const float* g1_Wih    = (const float*)d_in[26];
  const float* g1_Whh    = (const float*)d_in[27];
  const float* g1_bih    = (const float*)d_in[28];
  const float* g1_bhh    = (const float*)d_in[29];
  const float* lin2_W    = (const float*)d_in[30];
  const float* lin2_b    = (const float*)d_in[31];
  const int*   eidx      = (const int*)d_in[32];
  const int*   batch     = (const int*)d_in[33];
  const int* src = eidx;
  const int* dst = eidx + E;
  float* out = (float*)d_out;

  char* base = (char*)d_ws;
  size_t off = 0;
  auto allocB = [&](size_t bytes)->void*{
    off = (off + 15) & ~(size_t)15;
    void* p = base + off; off += bytes; return p;
  };
  h16* xpad16  = (h16*)allocB((size_t)N * 224 * 2);
  h16* xcat16  = (h16*)allocB((size_t)N * 256 * 2);
  h16* nbufA16 = (h16*)allocB((size_t)N * 128 * 2);
  h16* nbufB16 = (h16*)allocB((size_t)N * 128 * 2);
  h16* hbuf16  = (h16*)allocB((size_t)N * 128 * 2);
  h16* xt16    = (h16*)allocB((size_t)N * 128 * 2);
  h16* gi16    = (h16*)allocB((size_t)N * 384 * 2);
  h16* gh16    = (h16*)allocB((size_t)N * 384 * 2);
  h16* econ16  = (h16*)allocB((size_t)E * 160 * 2);   // reused as ye16 (E x 128)
  h16* xj16    = (h16*)allocB((size_t)E * 128 * 2);
  float* hbufF = (float*)allocB((size_t)N * 128 * 4);
  float* edot  = (float*)allocB((size_t)E * 4);
  float* attn  = (float*)allocB((size_t)E * 4);
  unsigned* segmax = (unsigned*)allocB((size_t)N * 4);
  float* segsum = (float*)allocB((size_t)N * 4);
  float* ndA   = (float*)allocB((size_t)N * 4);
  float* ndB   = (float*)allocB((size_t)N * 4);
  float* pool  = (float*)allocB((size_t)G * 128 * 4);
  h16* wt_emb  = (h16*)allocB((size_t)128 * 224 * 2);
  h16* wt_proj = (h16*)allocB((size_t)128 * 256 * 2);
  h16* wt_lin1 = (h16*)allocB((size_t)128 * 128 * 2);
  h16* wt_g1   = (h16*)allocB((size_t)128 * 160 * 2);
  h16* wt_g2   = (h16*)allocB((size_t)128 * 128 * 2);
  h16* wt_w0i  = (h16*)allocB((size_t)384 * 128 * 2);
  h16* wt_w0h  = (h16*)allocB((size_t)384 * 128 * 2);
  h16* wt_gat  = (h16*)allocB((size_t)128 * 128 * 2);
  h16* wt_w1i  = (h16*)allocB((size_t)384 * 128 * 2);
  h16* wt_w1h  = (h16*)allocB((size_t)384 * 128 * 2);
  h16* wt_lin2 = (h16*)allocB((size_t)128 * 128 * 2);

  const int T = 256;
  const int nN128 = N * 128, nE128 = E * 128;

  // ---- shared front end ----
  conv_pad_h<<<CEILDIV((long)N*224, T), T, 0, stream>>>(x, xpad16, 219, N, 224);
  conv_transpose_h<<<CEILDIV(128*224, T), T, 0, stream>>>(emb_W, wt_emb, 219, 128, 224);
  gemm(xpad16, wt_emb, emb_b, nbufA16, N, 128, 224, 224, 224, 128, 0, stream);
  bn_leaky_concat<<<CEILDIV(nN128, T), T, 0, stream>>>(nbufA16, x_motif, bn_gamma, bn_beta,
                                                       bn_mean, bn_var, xcat16, N);
  fill_f32<<<CEILDIV(G * 128, T), T, 0, stream>>>(pool, 0.f, G * 128);

  for (int h = 0; h < 4; ++h) {
    conv_transpose_h<<<CEILDIV(128*256, T), T, 0, stream>>>(proj_W + (size_t)h*256*128, wt_proj, 256, 128, 256);
    conv_transpose_h<<<CEILDIV(128*128, T), T, 0, stream>>>(lin1_W + (size_t)h*128*128, wt_lin1, 128, 128, 128);
    conv_transpose_h<<<CEILDIV(128*160, T), T, 0, stream>>>(gateL1 + (size_t)h*144*128, wt_g1, 144, 128, 160);
    conv_transpose_h<<<CEILDIV(128*128, T), T, 0, stream>>>(gateL2 + (size_t)h*128*128, wt_g2, 128, 128, 128);
    conv_pad_h<<<CEILDIV(384*128, T), T, 0, stream>>>(g0_Wih + (size_t)h*384*128, wt_w0i, 128, 384, 128);
    conv_pad_h<<<CEILDIV(384*128, T), T, 0, stream>>>(g0_Whh + (size_t)h*384*128, wt_w0h, 128, 384, 128);
    conv_transpose_h<<<CEILDIV(128*128, T), T, 0, stream>>>(gat_W + (size_t)h*128*128, wt_gat, 128, 128, 128);
    conv_pad_h<<<CEILDIV(384*128, T), T, 0, stream>>>(g1_Wih + (size_t)h*384*128, wt_w1i, 128, 384, 128);
    conv_pad_h<<<CEILDIV(384*128, T), T, 0, stream>>>(g1_Whh + (size_t)h*384*128, wt_w1h, 128, 384, 128);
    conv_transpose_h<<<CEILDIV(128*128, T), T, 0, stream>>>(lin2_W + (size_t)h*128*128, wt_lin2, 128, 128, 128);

    const float* pb  = proj_b    + (size_t)h * 128;
    const float* l1b = lin1_b    + (size_t)h * 128;
    const float* gal = gate_al   + (size_t)h * 128;
    const float* gar = gate_ar   + (size_t)h * 128;
    const float* gbi = gate_bias + (size_t)h * 128;
    const float* b0i = g0_bih    + (size_t)h * 384;
    const float* b0h = g0_bhh    + (size_t)h * 384;
    const float* gas = gat_as    + (size_t)h * 128;
    const float* gad = gat_ad    + (size_t)h * 128;
    const float* gbb = gat_bias  + (size_t)h * 128;
    const float* b1i = g1_bih    + (size_t)h * 384;
    const float* b1h = g1_bhh    + (size_t)h * 384;
    const float* l2b = lin2_b    + (size_t)h * 128;

    // project_motif + lin1(leaky) -> x in nbufA16
    gemm(xcat16, wt_proj, pb, nbufB16, N, 128, 256, 256, 256, 128, 0, stream);
    gemm(nbufB16, wt_lin1, l1b, nbufA16, N, 128, 128, 128, 128, 128, 1, stream);

    // ---- GATEConv ----
    edge_concat<<<CEILDIV(E * 160, T), T, 0, stream>>>(nbufA16, edge_attr, src, econ16, E);
    gemm(econ16, wt_g1, nullptr, xj16, E, 128, 160, 160, 160, 128, 1, stream);
    rowdot<<<CEILDIV(E, T), T, 0, stream>>>(xj16, gal, edot, E);
    rowdot<<<CEILDIV(N, T), T, 0, stream>>>(nbufA16, gar, ndA, N);
    combine_gate<<<CEILDIV(E, T), T, 0, stream>>>(edot, ndA, dst, E);
    fill_u32<<<CEILDIV(N, T), T, 0, stream>>>(segmax, 0u, N);
    fill_f32<<<CEILDIV(N, T), T, 0, stream>>>(segsum, 0.f, N);
    seg_max<<<CEILDIV(E, T), T, 0, stream>>>(edot, dst, segmax, E);
    seg_expsum<<<CEILDIV(E, T), T, 0, stream>>>(edot, dst, segmax, attn, segsum, E);
    seg_norm<<<CEILDIV(E, T), T, 0, stream>>>(attn, segsum, dst, E);
    gemm(xj16, wt_g2, nullptr, econ16 /*ye16*/, E, 128, 128, 128, 128, 128, 0, stream);
    fill_f32<<<CEILDIV(nN128, T), T, 0, stream>>>(hbufF, 0.f, nN128);
    scatter_wsum<<<CEILDIV(nE128, T), T, 0, stream>>>(econ16, attn, dst, hbufF, E);
    bias_elu_h<<<CEILDIV(nN128, T), T, 0, stream>>>(hbufF, gbi, hbuf16, nN128);

    // GRU0 -> x in nbufB16
    gemm(hbuf16,  wt_w0i, b0i, gi16, N, 384, 128, 128, 128, 384, 0, stream);
    gemm(nbufA16, wt_w0h, b0h, gh16, N, 384, 128, 128, 128, 384, 0, stream);
    gru_relu<<<CEILDIV(nN128, T), T, 0, stream>>>(gi16, gh16, nbufA16, nbufB16, nN128);

    // ---- GATConv ----
    gemm(nbufB16, wt_gat, nullptr, xt16, N, 128, 128, 128, 128, 128, 0, stream);
    rowdot<<<CEILDIV(N, T), T, 0, stream>>>(xt16, gas, ndB, N);
    rowdot<<<CEILDIV(N, T), T, 0, stream>>>(xt16, gad, ndA, N);
    combine_gat<<<CEILDIV(E, T), T, 0, stream>>>(ndB, ndA, src, dst, edot, E);
    fill_u32<<<CEILDIV(N, T), T, 0, stream>>>(segmax, 0u, N);
    fill_f32<<<CEILDIV(N, T), T, 0, stream>>>(segsum, 0.f, N);
    seg_max<<<CEILDIV(E, T), T, 0, stream>>>(edot, dst, segmax, E);
    seg_expsum<<<CEILDIV(E, T), T, 0, stream>>>(edot, dst, segmax, attn, segsum, E);
    seg_norm<<<CEILDIV(E, T), T, 0, stream>>>(attn, segsum, dst, E);
    fill_f32<<<CEILDIV(nN128, T), T, 0, stream>>>(hbufF, 0.f, nN128);
    scatter_gat<<<CEILDIV(nE128, T), T, 0, stream>>>(xt16, attn, src, dst, hbufF, E);
    bias_elu_h<<<CEILDIV(nN128, T), T, 0, stream>>>(hbufF, gbb, hbuf16, nN128);

    // GRU1 -> x in nbufA16
    gemm(hbuf16,  wt_w1i, b1i, gi16, N, 384, 128, 128, 128, 384, 0, stream);
    gemm(nbufB16, wt_w1h, b1h, gh16, N, 384, 128, 128, 128, 384, 0, stream);
    gru_relu<<<CEILDIV(nN128, T), T, 0, stream>>>(gi16, gh16, nbufB16, nbufA16, nN128);

    // lin2 + global_add_pool
    gemm(nbufA16, wt_lin2, l2b, hbuf16, N, 128, 128, 128, 128, 128, 0, stream);
    pool_add<<<CEILDIV(nN128, T), T, 0, stream>>>(hbuf16, batch, pool, nN128);
  }

  finalize_mean_relu<<<CEILDIV(G * 128, T), T, 0, stream>>>(pool, out, G * 128);
  (void)in_sizes; (void)n_in; (void)out_size; (void)ws_size;
}
